// MultiHeadSelfAttention_27419071217996
// MI455X (gfx1250) — matrix operand order repaired and checked by reading, not compiled, not run on hardware
//
#include <hip/hip_runtime.h>
#include <hip/hip_bf16.h>
#include <stdint.h>

// ---------- problem constants ----------
#define B_  2
#define T_  2048
#define C_  1024
#define H_  16
#define D_  64
#define BT_ (B_ * T_)          // 4096
#define C3_ (3 * C_)           // 3072

typedef _Float16 f16;
typedef __attribute__((ext_vector_type(16))) _Float16 v16h;
typedef __attribute__((ext_vector_type(8)))  _Float16 v8h;
typedef __attribute__((ext_vector_type(8)))  float    v8f;

__device__ __forceinline__ v16h cat8(v8h lo, v8h hi) {
    return __builtin_shufflevector(lo, hi, 0,1,2,3,4,5,6,7,8,9,10,11,12,13,14,15);
}
__device__ __forceinline__ v8h ldg8(const f16* p) {
    return *reinterpret_cast<const v8h*>(p);
}
__device__ __forceinline__ v8f wmma16(v16h a, v16h b, v8f c) {
    // D = A(16x32 f16) * B(32x16 f16) + C(16x16 f32)
    return __builtin_amdgcn_wmma_f32_16x16x32_f16(false, a, false, b, (short)0, c, false, false);
}

// CDNA5 async copy: global -> LDS, 16B per lane, tracked with ASYNCcnt.
__device__ __forceinline__ void async_b128(uint32_t lds_off, const f16* gptr) {
    asm volatile("global_load_async_to_lds_b128 %0, %1, off"
                 :: "v"(lds_off), "v"((unsigned long long)(uintptr_t)gptr)
                 : "memory");
}
__device__ __forceinline__ void wait_async_le(int pending) {
    if (pending) asm volatile("s_wait_asynccnt 4" ::: "memory");
    else         asm volatile("s_wait_asynccnt 0" ::: "memory");
}

// ---------- prep: f32 -> f16 convert ----------
__global__ void cvt_f32_f16_kernel(const float* __restrict__ in, f16* __restrict__ out, int n) {
    for (int i = blockIdx.x * blockDim.x + threadIdx.x; i < n; i += gridDim.x * blockDim.x)
        out[i] = (f16)in[i];
}

// ---------- prep: transpose [K][N] f32 -> [N][K] f16 ----------
__global__ void transpose_cvt_kernel(const float* __restrict__ in, f16* __restrict__ out,
                                     int K, int N) {
    int total = K * N;
    for (int i = blockIdx.x * blockDim.x + threadIdx.x; i < total; i += gridDim.x * blockDim.x) {
        int n = i / K;
        int k = i - n * K;
        out[(size_t)n * K + k] = (f16)in[(size_t)k * N + n];
    }
}

// ---------- QKV projection GEMM: qkv = xh @ W_qkv + b ----------
// Each wave: 32(M) x 64(N) tile; K loop unrolled 2x with ping-pong fragment
// sets so no register copies (and no WMMA->VALU WAR hazard nops) are needed.
__global__ void __launch_bounds__(128)
qkv_gemm_kernel(const f16* __restrict__ xh, const f16* __restrict__ WT,
                const float* __restrict__ bias,
                f16* __restrict__ qh, f16* __restrict__ kh, f16* __restrict__ vTh) {
    const int lane = threadIdx.x & 31;
    const int w    = threadIdx.x >> 5;
    const int ln   = lane & 15;
    const int half = lane >> 4;

    const int m0 = blockIdx.x * 32;                 // row tile in [0, BT)
    const int n0 = blockIdx.y * 256 + w * 64;       // col tile in [0, 3C)

    const f16* Ar[2] = { xh + (size_t)(m0 + ln) * C_,
                         xh + (size_t)(m0 + 16 + ln) * C_ };
    const f16* Br[4];
#pragma unroll
    for (int j = 0; j < 4; ++j) Br[j] = WT + (size_t)(n0 + 16 * j + ln) * C_;

    v8f acc[2][4] = {};
    v16h a0[2], b0v[4], a1[2], b1v[4];

    auto LDA = [&](v16h (&dst)[2], int k) {
#pragma unroll
        for (int i = 0; i < 2; ++i)
            dst[i] = cat8(ldg8(Ar[i] + k + 16 * half), ldg8(Ar[i] + k + 16 * half + 8));
    };
    auto LDB = [&](v16h (&dst)[4], int k) {
#pragma unroll
        for (int j = 0; j < 4; ++j)
            dst[j] = cat8(ldg8(Br[j] + k + 16 * half), ldg8(Br[j] + k + 16 * half + 8));
    };
    auto MMA = [&](v16h (&a)[2], v16h (&bv)[4]) {
#pragma unroll
        for (int i = 0; i < 2; ++i)
#pragma unroll
            for (int j = 0; j < 4; ++j)
                acc[i][j] = wmma16(a[i], bv[j], acc[i][j]);
    };

    LDA(a0, 0); LDB(b0v, 0);
    for (int k0 = 0; k0 < C_; k0 += 64) {
        __builtin_prefetch(Ar[0] + k0 + 160, 0, 1);
        LDA(a1, k0 + 32); LDB(b1v, k0 + 32);       // always in range (C_ % 64 == 0)
        MMA(a0, b0v);
        if (k0 + 64 < C_) { LDA(a0, k0 + 64); LDB(b0v, k0 + 64); }
        MMA(a1, b1v);
    }

#pragma unroll
    for (int j = 0; j < 4; ++j) {
        const int c = n0 + 16 * j + ln;             // output channel in [0, 3C)
        const int s = c >> 10;                      // 0=q 1=k 2=v
        const int h = (c >> 6) & (H_ - 1);
        const int d = c & (D_ - 1);
        const float bv = bias[c];
#pragma unroll
        for (int i = 0; i < 2; ++i)
#pragma unroll
            for (int r = 0; r < 8; ++r) {
                const int row = m0 + 16 * i + r + 8 * half;   // row = b*T + t
                const int b   = row >> 11;
                const int t   = row & (T_ - 1);
                const f16 hv = (f16)(acc[i][j][r] + bv);
                const size_t bh = (size_t)b * H_ + h;
                if (s == 0)       qh[(bh * T_ + t) * D_ + d] = hv;
                else if (s == 1)  kh[(bh * T_ + t) * D_ + d] = hv;
                else              vTh[(bh * D_ + d) * T_ + t] = hv;
            }
    }
}

// ---------- Flash attention ----------
// Block = 4 waves, same (b,h); K/V tiles staged in LDS via CDNA5 async copies,
// double-buffered; each wave owns 16 query rows.
__global__ void __launch_bounds__(128)
attention_kernel(const f16* __restrict__ qh, const f16* __restrict__ kh,
                 const f16* __restrict__ vTh, f16* __restrict__ ctx) {
    __shared__ __attribute__((aligned(128))) _Float16 Kb[2][32 * 64];   // [s][d]
    __shared__ __attribute__((aligned(128))) _Float16 Vb[2][64 * 32];   // [d][s]
    __shared__ __attribute__((aligned(128))) _Float16 Pbuf[4][16 * 32];

    const int tid  = threadIdx.x;
    const int lane = tid & 31;
    const int w    = tid >> 5;
    const int ln   = lane & 15;
    const int half = lane >> 4;

    const int h  = blockIdx.y;
    const int b  = blockIdx.z;
    const int t0 = blockIdx.x * 64 + w * 16;
    const size_t bh = (size_t)b * H_ + h;

    const f16* qrow  = qh  + (bh * T_ + t0 + ln) * D_;
    const f16* kbase = kh  + bh * T_ * D_;
    const f16* vbase = vTh + bh * D_ * T_;

    // cooperative async stage of K[s0:s0+32][0:64] and V^T[0:64][s0:s0+32]
    auto stage = [&](int buf, int s0) {
        const f16* ksrc = kbase + (size_t)s0 * D_;        // 4KB contiguous
        const uint32_t kdst = (uint32_t)(uintptr_t)(&Kb[buf][0]);
#pragma unroll
        for (int t = 0; t < 2; ++t) {
            const int c = tid + t * 128;                  // 256 chunks of 16B
            async_b128(kdst + c * 16, ksrc + c * 8);
        }
        const uint32_t vdst = (uint32_t)(uintptr_t)(&Vb[buf][0]);
#pragma unroll
        for (int t = 0; t < 2; ++t) {
            const int c   = tid + t * 128;                // d = c/4, col = (c%4)*8
            const int d   = c >> 2;
            const int col = (c & 3) * 8;
            async_b128(vdst + c * 16, vbase + (size_t)d * T_ + s0 + col);
        }
    };

    // q fragments (K-steps d=[0,32),[32,64)), fold 1/sqrt(D)=1/8 into q
    v16h aq[2];
#pragma unroll
    for (int kk = 0; kk < 2; ++kk) {
        v16h t = cat8(ldg8(qrow + 32 * kk + 16 * half),
                      ldg8(qrow + 32 * kk + 16 * half + 8));
        aq[kk] = t * (_Float16)0.125f;
    }

    float rowmax[8], rowsum[8];
    v8f O[4] = {};
#pragma unroll
    for (int r = 0; r < 8; ++r) { rowmax[r] = -3.0e38f; rowsum[r] = 0.0f; }

    stage(0, 0);
    stage(1, 32);

    for (int it = 0; it < T_ / 32; ++it) {
        const int s0  = it * 32;
        const int buf = it & 1;
        wait_async_le(s0 + 32 < T_);    // current tile resident; next may fly
        __syncthreads();

        // --- scores: two 16x16 tiles over keys [s0, s0+32) (B-frags from LDS) ---
        v8f S[2];
#pragma unroll
        for (int sc = 0; sc < 2; ++sc) {
            const f16* krow = &Kb[buf][(16 * sc + ln) * 64 + 16 * half];
            v16h b0 = cat8(*(const v8h*)(krow),      *(const v8h*)(krow + 8));
            v16h b1 = cat8(*(const v8h*)(krow + 32), *(const v8h*)(krow + 40));
            v8f z = {};
            z = wmma16(aq[0], b0, z);
            z = wmma16(aq[1], b1, z);
            S[sc] = z;
        }

        // --- online softmax (VGPR r, half -> row 8*half + r) ---
        float scl[8];
#pragma unroll
        for (int r = 0; r < 8; ++r) {
            float t = fmaxf(S[0][r], S[1][r]);
#pragma unroll
            for (int mask = 1; mask < 16; mask <<= 1)
                t = fmaxf(t, __shfl_xor(t, mask, 32));
            const float nm = fmaxf(rowmax[r], t);
            scl[r] = __expf(rowmax[r] - nm);
            rowmax[r] = nm;
        }
#pragma unroll
        for (int sc = 0; sc < 2; ++sc)
#pragma unroll
            for (int r = 0; r < 8; ++r) {
                const float p = __expf(S[sc][r] - rowmax[r]);
                S[sc][r] = p;
                Pbuf[w][(r + 8 * half) * 32 + 16 * sc + ln] = (f16)p;
            }
#pragma unroll
        for (int r = 0; r < 8; ++r) {
            float t = S[0][r] + S[1][r];
#pragma unroll
            for (int mask = 1; mask < 16; mask <<= 1)
                t += __shfl_xor(t, mask, 32);
            rowsum[r] = rowsum[r] * scl[r] + t;
        }
#pragma unroll
        for (int j = 0; j < 4; ++j)
#pragma unroll
            for (int r = 0; r < 8; ++r)
                O[j][r] *= scl[r];

        // P tile is wave-private in LDS; same-wave DS ops are in-order.
        asm volatile("" ::: "memory");

        const f16* prow = &Pbuf[w][ln * 32];
        v16h ap = cat8(*(const v8h*)(prow + 16 * half),
                       *(const v8h*)(prow + 16 * half + 8));
#pragma unroll
        for (int j = 0; j < 4; ++j) {
            const f16* vrow = &Vb[buf][(16 * j + ln) * 32 + 16 * half];
            v16h bv = cat8(*(const v8h*)(vrow), *(const v8h*)(vrow + 8));
            O[j] = wmma16(ap, bv, O[j]);
        }

        __syncthreads();                 // everyone done reading this buffer
        if (s0 + 64 < T_) stage(buf, s0 + 64);
    }

    // --- finalize: O /= rowsum, write ctx [B][T][C] f16 ---
    float inv[8];
#pragma unroll
    for (int r = 0; r < 8; ++r) inv[r] = 1.0f / rowsum[r];
#pragma unroll
    for (int j = 0; j < 4; ++j)
#pragma unroll
        for (int r = 0; r < 8; ++r) {
            const int t = t0 + r + 8 * half;
            const int d = 16 * j + ln;
            ctx[((size_t)b * T_ + t) * C_ + h * D_ + d] = (f16)(O[j][r] * inv[r]);
        }
}

// ---------- output projection: out = ctx @ W_out + b_out (f32 out) ----------
__global__ void __launch_bounds__(128)
outproj_gemm_kernel(const f16* __restrict__ ctx, const f16* __restrict__ WT,
                    const float* __restrict__ bias, float* __restrict__ out) {
    const int lane = threadIdx.x & 31;
    const int w    = threadIdx.x >> 5;
    const int ln   = lane & 15;
    const int half = lane >> 4;

    const int m0 = blockIdx.x * 32;
    const int n0 = blockIdx.y * 256 + w * 64;

    const f16* Ar[2] = { ctx + (size_t)(m0 + ln) * C_,
                         ctx + (size_t)(m0 + 16 + ln) * C_ };
    const f16* Br[4];
#pragma unroll
    for (int j = 0; j < 4; ++j) Br[j] = WT + (size_t)(n0 + 16 * j + ln) * C_;

    v8f acc[2][4] = {};
    v16h a0[2], b0v[4], a1[2], b1v[4];

    auto LDA = [&](v16h (&dst)[2], int k) {
#pragma unroll
        for (int i = 0; i < 2; ++i)
            dst[i] = cat8(ldg8(Ar[i] + k + 16 * half), ldg8(Ar[i] + k + 16 * half + 8));
    };
    auto LDB = [&](v16h (&dst)[4], int k) {
#pragma unroll
        for (int j = 0; j < 4; ++j)
            dst[j] = cat8(ldg8(Br[j] + k + 16 * half), ldg8(Br[j] + k + 16 * half + 8));
    };
    auto MMA = [&](v16h (&a)[2], v16h (&bv)[4]) {
#pragma unroll
        for (int i = 0; i < 2; ++i)
#pragma unroll
            for (int j = 0; j < 4; ++j)
                acc[i][j] = wmma16(a[i], bv[j], acc[i][j]);
    };

    LDA(a0, 0); LDB(b0v, 0);
    for (int k0 = 0; k0 < C_; k0 += 64) {
        __builtin_prefetch(Ar[0] + k0 + 160, 0, 1);
        LDA(a1, k0 + 32); LDB(b1v, k0 + 32);
        MMA(a0, b0v);
        if (k0 + 64 < C_) { LDA(a0, k0 + 64); LDB(b0v, k0 + 64); }
        MMA(a1, b1v);
    }

#pragma unroll
    for (int j = 0; j < 4; ++j) {
        const int c = n0 + 16 * j + ln;
        const float bv = bias[c];
#pragma unroll
        for (int i = 0; i < 2; ++i)
#pragma unroll
            for (int r = 0; r < 8; ++r) {
                const int row = m0 + 16 * i + r + 8 * half;
                out[(size_t)row * C_ + c] = acc[i][j][r] + bv;
            }
    }
}

// ---------- host launcher ----------
extern "C" void kernel_launch(void* const* d_in, const int* in_sizes, int n_in,
                              void* d_out, int out_size, void* d_ws, size_t ws_size,
                              hipStream_t stream) {
    const float* x     = (const float*)d_in[0];   // [B,T,C]
    const float* W_qkv = (const float*)d_in[1];   // [C,3C]
    const float* b_qkv = (const float*)d_in[2];   // [3C]
    const float* W_out = (const float*)d_in[3];   // [C,C]
    const float* b_out = (const float*)d_in[4];   // [C]
    float* out = (float*)d_out;                   // [B,T,C]

    char* ws = (char*)d_ws;
    size_t off = 0;
    auto take = [&](size_t elems) { f16* p = (f16*)(ws + off); off += elems * sizeof(f16); return p; };
    f16* xh    = take((size_t)BT_ * C_);           // 8 MB
    f16* WqkvT = take((size_t)C3_ * C_);           // 6 MB
    f16* WoutT = take((size_t)C_ * C_);            // 2 MB
    f16* qh    = take((size_t)B_ * H_ * T_ * D_);  // 8 MB
    f16* kh    = take((size_t)B_ * H_ * T_ * D_);  // 8 MB
    f16* vTh   = take((size_t)B_ * H_ * D_ * T_);  // 8 MB
    f16* ctx   = take((size_t)BT_ * C_);           // 8 MB (total 48 MB)
    (void)ws_size; (void)in_sizes; (void)n_in; (void)out_size;

    cvt_f32_f16_kernel<<<2048, 256, 0, stream>>>(x, xh, BT_ * C_);
    transpose_cvt_kernel<<<2048, 256, 0, stream>>>(W_qkv, WqkvT, C_, C3_);
    transpose_cvt_kernel<<<1024, 256, 0, stream>>>(W_out, WoutT, C_, C_);

    qkv_gemm_kernel<<<dim3(BT_ / 32, C3_ / 256), 128, 0, stream>>>(
        xh, WqkvT, b_qkv, qh, kh, vTh);

    attention_kernel<<<dim3(T_ / 64, H_, B_), 128, 0, stream>>>(qh, kh, vTh, ctx);

    outproj_gemm_kernel<<<dim3(BT_ / 32, C_ / 256), 128, 0, stream>>>(
        ctx, WoutT, b_out, out);
}
